// Attention_35905926595471
// MI455X (gfx1250) — matrix operand order repaired and checked by reading, not compiled, not run on hardware
//
#include <hip/hip_runtime.h>
#include <math.h>

typedef __attribute__((ext_vector_type(16))) __bf16 v16bf;
typedef __attribute__((ext_vector_type(8)))  __bf16 v8bf;
typedef __attribute__((ext_vector_type(8)))  float  v8f;

#define BQ   8
#define CDIM 192
#define NHW  16384          // 128*128
#define OQKV 576
#define HEADS 6
#define HD   32

// ---------------------------------------------------------------------------
// WMMA helpers (CDNA5 wave32, 16x16x32 bf16 -> f32)
// ---------------------------------------------------------------------------
__device__ inline v16bf make_v16(v8bf lo, v8bf hi) {
    v16bf r;
#pragma unroll
    for (int i = 0; i < 8; ++i) { r[i] = lo[i]; r[8 + i] = hi[i]; }
    return r;
}

// A-matrix fragment 16x32 bf16.  A is row-major [rows][rowstride], contraction
// along the row.  Lane L holds M = m0 + (L&15); K elems: {klo..klo+7, klo+16..klo+23}.
__device__ inline v16bf load_a_frag(const __bf16* A, int m0, long rowstride, long k0) {
    const int lane = threadIdx.x & 31;
    const int klo  = (lane >= 16) ? 8 : 0;
    const __bf16* p = A + (long)(m0 + (lane & 15)) * rowstride + k0 + klo;
    v8bf lo = *(const v8bf*)p;
    v8bf hi = *(const v8bf*)(p + 16);
    return make_v16(lo, hi);
}

// B-matrix fragment 32x16 bf16 where B[k][n] = Bm[n0+n][k0+k] (Bm row-major over k).
// Lane L holds N = n0 + (L&15); K elems {kb..kb+7, kb+16..kb+23}, kb = (L>=16)?8:0.
__device__ inline v16bf load_b_frag_rows(const __bf16* Bm, int n0, long rowstride, long k0) {
    const int lane = threadIdx.x & 31;
    const int kb   = (lane >= 16) ? 8 : 0;
    const __bf16* p = Bm + (long)(n0 + (lane & 15)) * rowstride + k0 + kb;
    v8bf lo = *(const v8bf*)p;
    v8bf hi = *(const v8bf*)(p + 16);
    return make_v16(lo, hi);
}

__device__ inline v8f wmma_bf16(v16bf a, v16bf b, v8f c) {
    return __builtin_amdgcn_wmma_f32_16x16x32_bf16(false, a, false, b, (short)0, c,
                                                   false, false);
}

// ---------------------------------------------------------------------------
// Prep kernels
// ---------------------------------------------------------------------------
__global__ void k_kv(const float* __restrict__ k_v, const float* __restrict__ w_kernel,
                     float* __restrict__ kv) {
    int i = blockIdx.x * blockDim.x + threadIdx.x;     // 8*384
    if (i >= BQ * 384) return;
    int b = i / 384, o = i % 384;
    float s = 0.f;
    for (int j = 0; j < 256; ++j) s += k_v[b * 256 + j] * w_kernel[o * 256 + j];
    kv[i] = s;
}

// Fold per-channel scale into qkv weights; turn per-channel bias into per-output bias.
__global__ void k_prepw(const float* __restrict__ w_qkv, const float* __restrict__ kv,
                        __bf16* __restrict__ Wq, float* __restrict__ bias) {
    int i = blockIdx.x * blockDim.x + threadIdx.x;     // 8*576
    if (i >= BQ * OQKV) return;
    int b = i / OQKV, o = i % OQKV;
    const float* kvb = kv + b * 384;
    float bs = 0.f;
    for (int c = 0; c < CDIM; ++c) {
        float w = w_qkv[o * CDIM + c];
        Wq[((long)b * OQKV + o) * CDIM + c] = (__bf16)(w * kvb[c]);
        bs += w * kvb[CDIM + c];
    }
    bias[i] = bs;
}

__global__ void k_f2bf(const float* __restrict__ src, __bf16* __restrict__ dst, long n) {
    long i = (long)blockIdx.x * blockDim.x + threadIdx.x;
    if (i < n) dst[i] = (__bf16)src[i];
}

// ---------------------------------------------------------------------------
// Batched GEMM: Out[b][o][n] = sum_c A[b][o][c] * X[b][c][n] (+ bias[b][o])
// Block: 128 threads (4 waves). Tile 64(M) x 64(N), K step 32.
// ---------------------------------------------------------------------------
template <typename OutT>
__global__ void k_gemm(const __bf16* __restrict__ A, long aBatchStride,
                       const float* __restrict__ bias,
                       const __bf16* __restrict__ X, OutT* __restrict__ Out,
                       int O, int K, int N) {
    const int b    = blockIdx.z;
    const int m0   = blockIdx.y * 64;
    const int n0   = blockIdx.x * 64;
    const int tid  = threadIdx.x;
    const int wave = tid >> 5;
    const int lane = tid & 31;

    __shared__ __bf16 ldsX[64][40];   // [n][k], padded stride 40 (80B, 16B-aligned rows)

    const __bf16* Ab = A + (long)b * aBatchStride;
    const __bf16* Xb = X + (long)b * K * N;

    v8f acc[4] = {};

    for (int k0 = 0; k0 < K; k0 += 32) {
        // stage X[k0:k0+32, n0:n0+64] transposed into LDS
        {
            int kk = tid >> 2;     // 0..31
            int cg = tid & 3;      // 0..3 -> 16 cols each
            const v8bf* src = (const v8bf*)(Xb + (long)(k0 + kk) * N + n0 + cg * 16);
            v8bf d0 = src[0];
            v8bf d1 = src[1];
#pragma unroll
            for (int j = 0; j < 8; ++j) {
                ldsX[cg * 16 + j][kk]     = d0[j];
                ldsX[cg * 16 + 8 + j][kk] = d1[j];
            }
        }
        __syncthreads();

        v16bf a = load_a_frag(Ab, m0 + wave * 16, K, k0);
#pragma unroll
        for (int ns = 0; ns < 4; ++ns) {
            v16bf bf = load_b_frag_rows(&ldsX[0][0], ns * 16, 40, 0);
            acc[ns] = wmma_bf16(a, bf, acc[ns]);
        }
        __syncthreads();
    }

    // epilogue: C/D layout -> lanes 0-15: M=r, lanes 16-31: M=8+r; N = lane&15
    const int mbase = m0 + wave * 16 + ((lane >= 16) ? 8 : 0);
    const int ncol  = n0 + (lane & 15);
#pragma unroll
    for (int ns = 0; ns < 4; ++ns) {
#pragma unroll
        for (int r = 0; r < 8; ++r) {
            int m = mbase + r;
            float v = acc[ns][r];
            if (bias) v += bias[(long)b * O + m];
            Out[((long)b * O + m) * N + ncol + ns * 16] = (OutT)v;
        }
    }
}

// ---------------------------------------------------------------------------
// Depthwise 3x3 conv, SAME padding, bf16 in/out, f32 accumulate
// ---------------------------------------------------------------------------
__global__ void k_dwconv(const __bf16* __restrict__ qkv, const float* __restrict__ w_dw,
                         __bf16* __restrict__ dw) {
    long i = (long)blockIdx.x * blockDim.x + threadIdx.x;
    if (i >= (long)BQ * OQKV * NHW) return;
    int  px = (int)(i & 127);
    int  py = (int)((i >> 7) & 127);
    long bc = i >> 14;
    int  ch = (int)(bc % OQKV);
    const float*  wt   = w_dw + ch * 9;
    const __bf16* base = qkv + (bc << 14);
    float s = 0.f;
#pragma unroll
    for (int dy = -1; dy <= 1; ++dy) {
        int yy = py + dy;
        if (yy < 0 || yy > 127) continue;
#pragma unroll
        for (int dx = -1; dx <= 1; ++dx) {
            int xx = px + dx;
            if (xx < 0 || xx > 127) continue;
            s += (float)base[yy * 128 + xx] * wt[(dy + 1) * 3 + (dx + 1)];
        }
    }
    dw[i] = (__bf16)s;
}

// ---------------------------------------------------------------------------
// L2 norms over the spatial axis for the q/k channels (first 384 channels)
// ---------------------------------------------------------------------------
__global__ void k_norm(const __bf16* __restrict__ dw, float* __restrict__ norminv) {
    int row = blockIdx.x;                       // b*384 + ch, ch < 384
    int b = row / 384, ch = row % 384;
    const __bf16* p = dw + ((long)b * OQKV + ch) * NHW;
    float s = 0.f;
    for (int i = threadIdx.x; i < NHW; i += 256) {
        float v = (float)p[i];
        s += v * v;
    }
    __shared__ float red[256];
    red[threadIdx.x] = s;
    __syncthreads();
    for (int off = 128; off > 0; off >>= 1) {
        if (threadIdx.x < off) red[threadIdx.x] += red[threadIdx.x + off];
        __syncthreads();
    }
    if (threadIdx.x == 0) norminv[row] = 1.0f / fmaxf(sqrtf(red[0]), 1e-12f);
}

__global__ void k_scale(const __bf16* __restrict__ dw, const float* __restrict__ norminv,
                        __bf16* __restrict__ qkvb) {
    long i = (long)blockIdx.x * blockDim.x + threadIdx.x;
    if (i >= (long)BQ * OQKV * NHW) return;
    long bc = i >> 14;
    int  b  = (int)(bc / OQKV);
    int  ch = (int)(bc % OQKV);
    float sc = (ch < 384) ? norminv[b * 384 + ch] : 1.0f;
    qkvb[i] = (__bf16)((float)dw[i] * sc);
}

// ---------------------------------------------------------------------------
// Attention logits + softmax: S = q·kT (32x32, K=16384), K sliced over 8 waves,
// LDS f32-atomic reduce, per-row softmax, write P (bf16 32x32 per (b,h)).
// ---------------------------------------------------------------------------
__global__ void k_attn_s(const __bf16* __restrict__ qkvb, const float* __restrict__ temp,
                         __bf16* __restrict__ P) {
    const int bh   = blockIdx.x;                 // 0..47
    const int b    = bh / HEADS, h = bh % HEADS;
    const int tid  = threadIdx.x;
    const int wave = tid >> 5;
    const int lane = tid & 31;

    const __bf16* q = qkvb + ((long)b * OQKV + h * HD) * NHW;
    const __bf16* k = qkvb + ((long)b * OQKV + CDIM + h * HD) * NHW;

    v8f acc[2][2] = {};
    const long nbase = (long)wave * (NHW / 8);
    for (int step = 0; step < (NHW / 8) / 32; ++step) {
        long k0 = nbase + (long)step * 32;
        v16bf a0 = load_a_frag(q, 0, NHW, k0);
        v16bf a1 = load_a_frag(q, 16, NHW, k0);
        v16bf b0 = load_b_frag_rows(k, 0, NHW, k0);
        v16bf b1 = load_b_frag_rows(k, 16, NHW, k0);
        acc[0][0] = wmma_bf16(a0, b0, acc[0][0]);
        acc[0][1] = wmma_bf16(a0, b1, acc[0][1]);
        acc[1][0] = wmma_bf16(a1, b0, acc[1][0]);
        acc[1][1] = wmma_bf16(a1, b1, acc[1][1]);
    }

    __shared__ float S[32][32];
    for (int i = tid; i < 1024; i += 256) ((float*)S)[i] = 0.f;
    __syncthreads();

    const int mo = (lane >= 16) ? 8 : 0;
    const int d  = lane & 15;
#pragma unroll
    for (int mt = 0; mt < 2; ++mt)
#pragma unroll
        for (int nt = 0; nt < 2; ++nt)
#pragma unroll
            for (int r = 0; r < 8; ++r)
                atomicAdd(&S[mt * 16 + mo + r][nt * 16 + d], acc[mt][nt][r]);
    __syncthreads();

    if (tid < 32) {
        float t  = temp[h];
        float rowv[32];
        float mx = -1e30f;
#pragma unroll
        for (int j = 0; j < 32; ++j) {
            float v = S[tid][j] * t;
            rowv[j] = v;
            mx = fmaxf(mx, v);
        }
        float sum = 0.f;
#pragma unroll
        for (int j = 0; j < 32; ++j) {
            rowv[j] = __expf(rowv[j] - mx);
            sum += rowv[j];
        }
        float inv = 1.0f / sum;
        __bf16* Prow = P + (long)bh * 1024 + tid * 32;
#pragma unroll
        for (int j = 0; j < 32; ++j) Prow[j] = (__bf16)(rowv[j] * inv);
    }
}

// ---------------------------------------------------------------------------
// out = P(32x32) @ v(32x16384). Block: 128 threads / 4 waves; tile 32(M)x64(N),
// single K=32 WMMA step; v tile staged transposed in LDS.
// ---------------------------------------------------------------------------
__global__ void k_attn_o(const __bf16* __restrict__ qkvb, const __bf16* __restrict__ P,
                         __bf16* __restrict__ attnout) {
    const int bh   = blockIdx.y;
    const int b    = bh / HEADS, h = bh % HEADS;
    const int n0   = blockIdx.x * 64;
    const int tid  = threadIdx.x;
    const int wave = tid >> 5;
    const int lane = tid & 31;

    const __bf16* v = qkvb + ((long)b * OQKV + 2 * CDIM + h * HD) * NHW;

    __shared__ __bf16 vT[64][40];
    {
        int dd = tid >> 2;     // 0..31 (channel d)
        int cg = tid & 3;      // 0..3
        const v8bf* src = (const v8bf*)(v + (long)dd * NHW + n0 + cg * 16);
        v8bf d0 = src[0];
        v8bf d1 = src[1];
#pragma unroll
        for (int j = 0; j < 8; ++j) {
            vT[cg * 16 + j][dd]     = d0[j];
            vT[cg * 16 + 8 + j][dd] = d1[j];
        }
    }
    __syncthreads();

    const __bf16* Pb = P + (long)bh * 1024;
    v16bf a0 = load_a_frag(Pb, 0, 32, 0);
    v16bf a1 = load_a_frag(Pb, 16, 32, 0);
    v16bf bf = load_b_frag_rows(&vT[0][0], wave * 16, 40, 0);
    v8f acc0 = {}, acc1 = {};
    acc0 = wmma_bf16(a0, bf, acc0);
    acc1 = wmma_bf16(a1, bf, acc1);

    const int mo = (lane >= 16) ? 8 : 0;
    const int nc = n0 + wave * 16 + (lane & 15);
#pragma unroll
    for (int r = 0; r < 8; ++r) {
        attnout[((long)b * CDIM + h * HD + mo + r) * NHW + nc]      = (__bf16)acc0[r];
        attnout[((long)b * CDIM + h * HD + 16 + mo + r) * NHW + nc] = (__bf16)acc1[r];
    }
}

// ---------------------------------------------------------------------------
// Host launch
// ---------------------------------------------------------------------------
extern "C" void kernel_launch(void* const* d_in, const int* in_sizes, int n_in,
                              void* d_out, int out_size, void* d_ws, size_t ws_size,
                              hipStream_t stream) {
    const float* x           = (const float*)d_in[0];
    const float* k_v         = (const float*)d_in[1];
    const float* w_kernel    = (const float*)d_in[2];
    const float* w_qkv       = (const float*)d_in[3];
    const float* w_dw        = (const float*)d_in[4];
    const float* w_proj      = (const float*)d_in[5];
    const float* temperature = (const float*)d_in[6];
    float* out = (float*)d_out;

    char*  ws  = (char*)d_ws;
    size_t off = 0;
    auto alloc = [&](size_t bytes) -> void* {
        off = (off + 255) & ~(size_t)255;
        void* p = ws + off;
        off += bytes;
        return p;
    };

    float*  kv      = (float*)alloc((size_t)BQ * 384 * 4);
    float*  biasq   = (float*)alloc((size_t)BQ * OQKV * 4);
    __bf16* Wqb     = (__bf16*)alloc((size_t)BQ * OQKV * CDIM * 2);
    __bf16* wprojb  = (__bf16*)alloc((size_t)CDIM * CDIM * 2);
    float*  norminv = (float*)alloc((size_t)BQ * 384 * 4);
    __bf16* P       = (__bf16*)alloc((size_t)BQ * HEADS * 32 * 32 * 2);
    __bf16* xbuf    = (__bf16*)alloc((size_t)BQ * CDIM * NHW * 2);   // xb, later attnout
    __bf16* qkvbuf  = (__bf16*)alloc((size_t)BQ * OQKV * NHW * 2);   // qkv, later qkvb
    __bf16* dwbuf   = (__bf16*)alloc((size_t)BQ * OQKV * NHW * 2);

    const long nX  = (long)BQ * CDIM * NHW;    // 25165824
    const long nQK = (long)BQ * OQKV * NHW;    // 75497472

    // 0. kv = k_v @ w_kernel.T ; folded weights/bias ; wproj -> bf16
    k_kv<<<12, 256, 0, stream>>>(k_v, w_kernel, kv);
    k_prepw<<<18, 256, 0, stream>>>(w_qkv, kv, Wqb, biasq);
    k_f2bf<<<(CDIM * CDIM) / 256, 256, 0, stream>>>(w_proj, wprojb, CDIM * CDIM);

    // 1. x -> bf16
    k_f2bf<<<(unsigned)(nX / 256), 256, 0, stream>>>(x, xbuf, nX);

    // 2. qkv GEMM (WMMA): per-batch (576x192)@(192x16384) + bias, bf16 out
    k_gemm<__bf16><<<dim3(NHW / 64, OQKV / 64, BQ), 128, 0, stream>>>(
        Wqb, (long)OQKV * CDIM, biasq, xbuf, qkvbuf, OQKV, CDIM, NHW);

    // 3. depthwise 3x3
    k_dwconv<<<(unsigned)(nQK / 256), 256, 0, stream>>>(qkvbuf, w_dw, dwbuf);

    // 4. L2 norms for q,k rows; 5. normalize + repack (overwrites qkvbuf)
    k_norm<<<BQ * 384, 256, 0, stream>>>(dwbuf, norminv);
    k_scale<<<(unsigned)(nQK / 256), 256, 0, stream>>>(dwbuf, norminv, qkvbuf);

    // 6. logits + softmax (WMMA, K sliced over 8 waves)
    k_attn_s<<<BQ * HEADS, 256, 0, stream>>>(qkvbuf, temperature, P);

    // 7. out = P @ v (WMMA), write into xbuf (xb is dead)
    k_attn_o<<<dim3(NHW / 64, BQ * HEADS), 128, 0, stream>>>(qkvbuf, P, xbuf);

    // 8. final projection (WMMA): per-batch (192x192)@(192x16384) -> f32 d_out
    k_gemm<float><<<dim3(NHW / 64, CDIM / 64, BQ), 128, 0, stream>>>(
        wprojb, 0, nullptr, xbuf, out, CDIM, CDIM, NHW);

    (void)in_sizes; (void)n_in; (void)out_size; (void)ws_size;
}
